// LiquidNeuralEngine_48335561949510
// MI455X (gfx1250) — compile-verified, matrix operand order checked by reading, not run on hardware
//
#include <hip/hip_runtime.h>

#define BATCH   32768
#define DH      512
#define DIN     512
#define DOUT    256
#define STEPS   20
#define MTILE   64
#define HSTR    520      // bf16 elements per LDS row (1040 B, 16B aligned, bank-conflict-free)
#define NWAVES  16
#define NTHREADS 512

typedef __attribute__((ext_vector_type(16))) __bf16 v16bf;
typedef __attribute__((ext_vector_type(8)))  __bf16 v8bf;
typedef __attribute__((ext_vector_type(8)))  float  v8f;

union FragU { v16bf v; v8bf h[2]; };

// ---------------------------------------------------------------------------
// Weight pre-pack: f32 row-major [K][N] -> bf16 per-lane WMMA B-fragment layout.
// Packed element index = ((kt*(N/16)+nt)*32 + lane)*16 + e, where for lane l,
// element e: v=e/2, p=e&1, grp=l/16, and
//   K = kt*32 + (v>=4 ? 16 : 0) + grp*8 + (v&3)*2 + p,  n = nt*16 + (l&15)
// (matches the CDNA5 ISA 16-bit matrix VGPR striping).
// ---------------------------------------------------------------------------
__global__ void pack_w_bf16(const float* __restrict__ W, __bf16* __restrict__ out,
                            int K, int N) {
  int idx = blockIdx.x * blockDim.x + threadIdx.x;
  if (idx >= K * N) return;
  int e    = idx & 15;
  int lane = (idx >> 4) & 31;
  int tile = idx >> 9;
  int ntn  = N >> 4;
  int kt   = tile / ntn;
  int nt   = tile - kt * ntn;
  int grp  = lane >> 4;
  int n    = nt * 16 + (lane & 15);
  int v    = e >> 1, p = e & 1;
  int k    = kt * 32 + ((v >= 4) ? 16 : 0) + grp * 8 + ((v & 3) * 2) + p;
  out[idx] = (__bf16)W[(size_t)k * N + n];
}

__device__ __forceinline__ float fast_tanh(float x) {
  // tanh(x) = 1 - 2/(exp(2x)+1); saturates correctly at +/-1 for large |x|
  float e = __expf(2.0f * x);
  return 1.0f - 2.0f * __builtin_amdgcn_rcpf(e + 1.0f);
}

// Shared memory: hbuf MUST be at offset 0 so every A-fragment ds_load byte
// offset (<= ~51KB) fits the 16-bit DS immediate field (single base VGPR).
struct __align__(64) SMem {
  __bf16 hbuf[MTILE * HSTR];            // 66,560 B @ offset 0
  float  xinbuf[NWAVES * 8 * 32 * 8];   // 131,072 B @ offset 66,560 (32B aligned)
};

// GEMM over a 64xK bf16 tile in LDS (row-major, stride HSTR) times packed
// weights. Each wave computes 4 M-tiles x NT N-tiles, accumulating f32.
// B fragments loaded first (per kt), then each A fragment feeds NT WMMAs.
template <int NT>
__device__ __forceinline__ void gemm16(const __bf16* __restrict__ hb,
                                       const __bf16* __restrict__ W,
                                       int nTilesN, int ntile0,
                                       v8f (&acc)[4][NT], int lane) {
  const int lm  = lane & 15;
  const int grp = lane >> 4;
  const __bf16* abase = hb + lm * HSTR + grp * 8;  // one base; rest immediates
  for (int kt = 0; kt < 16; ++kt) {
    FragU b[NT];
#pragma unroll
    for (int nt = 0; nt < NT; ++nt) {
      const __bf16* wp =
          W + ((size_t)(kt * nTilesN + ntile0 + nt) * 32 + lane) * 16;
      b[nt].h[0] = *(const v8bf*)wp;
      b[nt].h[1] = *(const v8bf*)(wp + 8);
    }
#pragma unroll
    for (int mt = 0; mt < 4; ++mt) {
      const __bf16* p = abase + mt * 16 * HSTR + kt * 32;
      FragU a;
      a.h[0] = *(const v8bf*)p;         // K block {0..7}/{8..15}
      a.h[1] = *(const v8bf*)(p + 16);  // K block {16..23}/{24..31}
#pragma unroll
      for (int nt = 0; nt < NT; ++nt)
        acc[mt][nt] = __builtin_amdgcn_wmma_f32_16x16x32_bf16(
            false, a.v, false, b[nt].v, (short)0, acc[mt][nt], false, false);
    }
  }
}

__launch_bounds__(NTHREADS)
__global__ void liquid_fused(const float* __restrict__ features,
                             const __bf16* __restrict__ Wb_in,
                             const __bf16* __restrict__ Wb_h,
                             const __bf16* __restrict__ Wb_2,
                             const __bf16* __restrict__ Wb_out,
                             const float* __restrict__ b_h,
                             const float* __restrict__ tau,
                             const float* __restrict__ b2,
                             const float* __restrict__ b_out,
                             float* __restrict__ out) {
  __shared__ SMem sm;

  const int tid  = threadIdx.x;
  const int wave = tid >> 5;
  const int lane = tid & 31;
  const int lm   = lane & 15;
  const int grp  = lane >> 4;
  const int row0 = blockIdx.x * MTILE;
  const int n0   = wave * 32;   // wave's 32-column strip (2 n-tiles)

  // ---- phase 0: features tile -> bf16 LDS (row-major) ----
  for (int i = tid; i < MTILE * DIN; i += NTHREADS) {
    int r = i >> 9, c = i & (DIN - 1);
    sm.hbuf[r * HSTR + c] = (__bf16)features[(size_t)(row0 + r) * DIN + c];
  }
  __syncthreads();

  const v8f vzero = {0.f, 0.f, 0.f, 0.f, 0.f, 0.f, 0.f, 0.f};

  // ---- phase 1: xin = X @ W_in + b_h  (kept in LDS, wave-private layout) ----
  v8f g[4][2];
#pragma unroll
  for (int mt = 0; mt < 4; ++mt)
#pragma unroll
    for (int nt = 0; nt < 2; ++nt) g[mt][nt] = vzero;

  gemm16<2>(sm.hbuf, Wb_in, DH >> 4, wave * 2, g, lane);

  float rcptau[2];
#pragma unroll
  for (int nt = 0; nt < 2; ++nt)
    rcptau[nt] = __builtin_amdgcn_rcpf(tau[n0 + nt * 16 + lm]);

  const float dt = 1.0f / (float)STEPS;

  // store xin; step 0 folded in: h0 = 0  ->  h1 = dt * tanh(xin)
  float* xin = &sm.xinbuf[wave * 2048];
  v8f h[4][2];
#pragma unroll
  for (int nt = 0; nt < 2; ++nt) {
    float bh = b_h[n0 + nt * 16 + lm];
#pragma unroll
    for (int mt = 0; mt < 4; ++mt) {
      float* xp = xin + ((mt * 2 + nt) * 32 + lane) * 8;
#pragma unroll
      for (int r = 0; r < 8; ++r) {
        float xv = g[mt][nt][r] + bh;
        xp[r] = xv;
        h[mt][nt][r] = dt * fast_tanh(xv);
      }
    }
  }

  // ---- phase 2: remaining 19 Euler steps of the liquid recurrence ----
#pragma unroll 1
  for (int step = 1; step < STEPS; ++step) {
    // h (f32 regs, C-fragment layout) -> bf16 row-major LDS
#pragma unroll
    for (int mt = 0; mt < 4; ++mt)
#pragma unroll
      for (int nt = 0; nt < 2; ++nt) {
        int nn = n0 + nt * 16 + lm;
#pragma unroll
        for (int r = 0; r < 8; ++r)
          sm.hbuf[(mt * 16 + grp * 8 + r) * HSTR + nn] = (__bf16)h[mt][nt][r];
      }
    __syncthreads();

    // g starts as xin, WMMA accumulates h @ W_h on top
#pragma unroll
    for (int mt = 0; mt < 4; ++mt)
#pragma unroll
      for (int nt = 0; nt < 2; ++nt)
        g[mt][nt] = *(const v8f*)(xin + ((mt * 2 + nt) * 32 + lane) * 8);

    gemm16<2>(sm.hbuf, Wb_h, DH >> 4, wave * 2, g, lane);
    __syncthreads();

    // Euler: h += dt * (tanh(g) - h/tau)
#pragma unroll
    for (int mt = 0; mt < 4; ++mt)
#pragma unroll
      for (int nt = 0; nt < 2; ++nt)
#pragma unroll
        for (int r = 0; r < 8; ++r) {
          float hv = h[mt][nt][r];
          float tv = fast_tanh(g[mt][nt][r]);
          h[mt][nt][r] = hv + dt * (tv - hv * rcptau[nt]);
        }
  }

  // ---- phase 3: a = tanh(h @ W2 + b2) ----
#pragma unroll
  for (int mt = 0; mt < 4; ++mt)
#pragma unroll
    for (int nt = 0; nt < 2; ++nt) {
      int nn = n0 + nt * 16 + lm;
#pragma unroll
      for (int r = 0; r < 8; ++r)
        sm.hbuf[(mt * 16 + grp * 8 + r) * HSTR + nn] = (__bf16)h[mt][nt][r];
    }
  __syncthreads();

#pragma unroll
  for (int mt = 0; mt < 4; ++mt)
#pragma unroll
    for (int nt = 0; nt < 2; ++nt) g[mt][nt] = vzero;
  gemm16<2>(sm.hbuf, Wb_2, DH >> 4, wave * 2, g, lane);
  __syncthreads();

#pragma unroll
  for (int nt = 0; nt < 2; ++nt) {
    float bb = b2[n0 + nt * 16 + lm];
    int nn = n0 + nt * 16 + lm;
#pragma unroll
    for (int mt = 0; mt < 4; ++mt)
#pragma unroll
      for (int r = 0; r < 8; ++r)
        sm.hbuf[(mt * 16 + grp * 8 + r) * HSTR + nn] =
            (__bf16)fast_tanh(g[mt][nt][r] + bb);
  }
  __syncthreads();

  // ---- phase 4: out = a @ W_out + b_out (N=256 -> 1 n-tile per wave) ----
  v8f o[4][1];
#pragma unroll
  for (int mt = 0; mt < 4; ++mt) o[mt][0] = vzero;

  gemm16<1>(sm.hbuf, Wb_out, DOUT >> 4, wave, o, lane);

  {
    int nn = wave * 16 + lm;
    float bo = b_out[nn];
#pragma unroll
    for (int mt = 0; mt < 4; ++mt)
#pragma unroll
      for (int r = 0; r < 8; ++r)
        out[(size_t)(row0 + mt * 16 + grp * 8 + r) * DOUT + nn] =
            o[mt][0][r] + bo;
  }
}

extern "C" void kernel_launch(void* const* d_in, const int* in_sizes, int n_in,
                              void* d_out, int out_size, void* d_ws, size_t ws_size,
                              hipStream_t stream) {
  (void)in_sizes; (void)n_in; (void)out_size; (void)ws_size;
  const float* features = (const float*)d_in[0];
  const float* W_in  = (const float*)d_in[1];
  const float* W_h   = (const float*)d_in[2];
  const float* b_h   = (const float*)d_in[3];
  const float* tau   = (const float*)d_in[4];
  const float* W2    = (const float*)d_in[5];
  const float* b2    = (const float*)d_in[6];
  const float* W_out = (const float*)d_in[7];
  const float* b_out = (const float*)d_in[8];
  float* out = (float*)d_out;

  char* ws = (char*)d_ws;
  __bf16* Wb_in  = (__bf16*)(ws + (size_t)0);
  __bf16* Wb_h   = (__bf16*)(ws + (size_t)512 * 512 * 2);
  __bf16* Wb_2   = (__bf16*)(ws + (size_t)2 * 512 * 512 * 2);
  __bf16* Wb_out = (__bf16*)(ws + (size_t)3 * 512 * 512 * 2);

  const int nsq = 512 * 512, nro = 512 * 256;
  pack_w_bf16<<<(nsq + 255) / 256, 256, 0, stream>>>(W_in,  Wb_in,  512, 512);
  pack_w_bf16<<<(nsq + 255) / 256, 256, 0, stream>>>(W_h,   Wb_h,   512, 512);
  pack_w_bf16<<<(nsq + 255) / 256, 256, 0, stream>>>(W2,    Wb_2,   512, 512);
  pack_w_bf16<<<(nro + 255) / 256, 256, 0, stream>>>(W_out, Wb_out, 512, 256);

  liquid_fused<<<BATCH / MTILE, NTHREADS, 0, stream>>>(
      features, Wb_in, Wb_h, Wb_2, Wb_out, b_h, tau, b2, b_out, out);
}
